// BahdanauAttention_45449343926306
// MI455X (gfx1250) — matrix operand order repaired and checked by reading, not compiled, not run on hardware
//
#include <hip/hip_runtime.h>
#include <hip/hip_bf16.h>

typedef __attribute__((ext_vector_type(16))) _Float16 v16h;
typedef __attribute__((ext_vector_type(8)))  _Float16 v8h;
typedef __attribute__((ext_vector_type(8)))  float    v8f;

// Problem sizes (fixed by the reference)
static constexpr int Bn = 64;
static constexpr int Tn = 2048;
static constexpr int Fn = 512;
static constexpr int Un = 512;

// ---------------------------------------------------------------------------
// Kernel 1: pack W1 (fp32, [F,U] row-major) into f16 WMMA B-fragment layout.
// Layout: w1p[kt][nt][lane][16 f16], kt in [0,16), nt in [0,32), lane in [0,32).
//   kt-major so a wave's 4 consecutive nt tiles are contiguous (immediate
//   offsets in global_load_b128) and the kt advance is one scalar add (32 KB).
// Per ISA 7.12.2 (16-bit B 32x16): lanes 0-15 hold K=0..7,16..23, lanes 16-31 hold
// K=8..15,24..31; within a lane, VGPR v holds rows K=8g+2(v&3)+16(v>>2) (lo) / +1 (hi),
// column N = lane & 15.
// ---------------------------------------------------------------------------
__global__ __launch_bounds__(256)
void pack_w1_kernel(const float* __restrict__ W1, _Float16* __restrict__ w1p)
{
    int idx  = blockIdx.x * 256 + threadIdx.x;     // < 512*512 = 2^18
    int i    = idx & 15;                           // element within lane chunk
    int lane = (idx >> 4) & 31;
    int nt   = (idx >> 9) & 31;                    // N-tile (16 wide)
    int kt   = idx >> 14;                          // K-tile (32 wide)
    int g  = lane >> 4;                            // lane group: K halves
    int n  = lane & 15;                            // column
    int v  = i >> 1;                               // VGPR index
    int hh = i & 1;                                // lo/hi half
    int k  = kt * 32 + 8 * g + (i & 7) + 16 * (i >> 3);
    int u  = nt * 16 + n;
    w1p[idx] = (_Float16)W1[(size_t)k * Un + u];
}

// ---------------------------------------------------------------------------
// Kernel 2: projh[b,u] = hidden[b,:] @ W2[:,u] + b2[u]   (tiny GEMM, FMA)
// ---------------------------------------------------------------------------
__global__ __launch_bounds__(256)
void projh_kernel(const float* __restrict__ hidden, const float* __restrict__ W2,
                  const float* __restrict__ b2, float* __restrict__ projh)
{
    __shared__ float h[Fn];
    const int b = blockIdx.x, tid = threadIdx.x;
    h[tid]       = hidden[b * Fn + tid];
    h[tid + 256] = hidden[b * Fn + tid + 256];
    __syncthreads();
#pragma unroll
    for (int uu = 0; uu < 2; ++uu) {
        const int u = tid + uu * 256;
        float acc = b2[u];
#pragma unroll 8
        for (int f = 0; f < Fn; ++f)
            acc = fmaf(h[f], W2[(size_t)f * Un + u], acc);
        projh[b * Un + u] = acc;
    }
}

// ---------------------------------------------------------------------------
// Kernel 3 (main): per (b, 32-row t-tile)
//   score[b,t] = sum_u tanh( (features[b,t,:] @ W1)[u] + b1[u] + projh[b,u] ) * V[u]
// 8 waves/block; each wave owns 2 M-tiles x 4 N-tiles.  Each B fragment feeds
// two independent v_wmma_f32_16x16x32_f16 (halves L2 B-traffic per FLOP and
// hides load latency).  A-tile (32x512) staged in LDS as f16.  Row scores
// reduced across columns/waves via ds_add_f32 LDS atomics.
// ---------------------------------------------------------------------------
__global__ __launch_bounds__(256)
void score_kernel(const float* __restrict__ features,
                  const _Float16* __restrict__ w1p,
                  const float* __restrict__ b1,
                  const float* __restrict__ projh,
                  const float* __restrict__ V,
                  float* __restrict__ score)
{
    constexpr int MROWS = 32;
    constexpr int LDS_STRIDE = Fn + 16;            // pad: keeps 16B align, rotates banks
    __shared__ __align__(16) _Float16 As[MROWS * LDS_STRIDE];
    __shared__ float sscore[MROWS];

    const int tid  = threadIdx.x;
    const int lane = tid & 31;
    const int wave = tid >> 5;
    const int b    = blockIdx.x >> 6;              // / (T/32)
    const int t0   = (blockIdx.x & 63) << 5;

    // Stage 32 x 512 feature rows, fp32 -> f16, into LDS.
    const float* frow = features + ((size_t)b * Tn + t0) * Fn;
    for (int i = tid; i < MROWS * Fn; i += 256) {
        const int r = i >> 9;                      // / 512
        const int c = i & (Fn - 1);
        As[r * LDS_STRIDE + c] = (_Float16)frow[(size_t)r * Fn + c];
    }
    if (tid < MROWS) sscore[tid] = 0.0f;
    __syncthreads();

    // A-fragment addressing (16-bit A 16x32, ISA 7.12.2): lane L -> row M=L&15;
    // lanes 0-15 carry K = kb+0..7 and kb+16..23; lanes 16-31 carry +8.
    const int koff = (lane >> 4) << 3;             // 0 or 8
    const _Float16* arow0 = &As[(lane & 15) * LDS_STRIDE];        // M-tile 0
    const _Float16* arow1 = arow0 + 16 * LDS_STRIDE;              // M-tile 1

    // B base for this wave at kt=0: tiles nt = wave*4 .. wave*4+3 are the next
    // 4 KB; kt advance = 32*32*16 halves = 32 KB (scalar-uniform stride).
    const _Float16* bbase = w1p + (((size_t)(wave << 2) * 32 + lane) << 4);
    constexpr int KT_STRIDE = 32 * 32 * 16;        // halves per kt step
    constexpr int NT_STRIDE = 32 * 16;             // halves per nt step (512)

    v8f acc[2][4] = {};                            // [mtile][ntile] f32 accum

    for (int kt = 0; kt < 16; ++kt) {
        const int kb = kt * 32;
        union { v16h v; v8h h[2]; } a0, a1;
        a0.h[0] = *(const v8h*)(arow0 + kb + koff);
        a0.h[1] = *(const v8h*)(arow0 + kb + 16 + koff);
        a1.h[0] = *(const v8h*)(arow1 + kb + koff);
        a1.h[1] = *(const v8h*)(arow1 + kb + 16 + koff);
        const _Float16* bk = bbase + (size_t)kt * KT_STRIDE;
#pragma unroll
        for (int j = 0; j < 4; ++j) {
            union { v16h v; v8h h[2]; } bb;
            bb.h[0] = *(const v8h*)(bk + j * NT_STRIDE);
            bb.h[1] = *(const v8h*)(bk + j * NT_STRIDE + 8);
            acc[0][j] = __builtin_amdgcn_wmma_f32_16x16x32_f16(
                false, a0.v, false, bb.v, (short)0, acc[0][j], false, false);
            acc[1][j] = __builtin_amdgcn_wmma_f32_16x16x32_f16(
                false, a1.v, false, bb.v, (short)0, acc[1][j], false, false);
        }
    }

    // Epilogue: C/D layout -> lane's element r is row M = r + 8*(lane>>4),
    // column N = lane&15.  Fuse bias + projh + tanh + dot-with-V.
    float part[2][8];
#pragma unroll
    for (int mt = 0; mt < 2; ++mt)
#pragma unroll
        for (int r = 0; r < 8; ++r) part[mt][r] = 0.0f;
#pragma unroll
    for (int j = 0; j < 4; ++j) {
        const int u      = (((wave << 2) + j) << 4) + (lane & 15);
        const float bias = b1[u] + projh[b * Un + u];
        const float vv   = V[u];
#pragma unroll
        for (int mt = 0; mt < 2; ++mt)
#pragma unroll
            for (int r = 0; r < 8; ++r)
                part[mt][r] += tanhf(acc[mt][j][r] + bias) * vv;
    }
    const int mbase = (lane >> 4) << 3;
#pragma unroll
    for (int mt = 0; mt < 2; ++mt)
#pragma unroll
        for (int r = 0; r < 8; ++r)
            atomicAdd(&sscore[mt * 16 + mbase + r], part[mt][r]);   // ds_add_f32
    __syncthreads();
    if (tid < MROWS) score[(size_t)b * Tn + t0 + tid] = sscore[tid];
}

// ---------------------------------------------------------------------------
// Kernel 4: softmax over T per batch.  (bv omitted: softmax is shift-invariant.)
// ---------------------------------------------------------------------------
__global__ __launch_bounds__(256)
void softmax_kernel(const float* __restrict__ score, float* __restrict__ attn)
{
    __shared__ float red[256];
    const int b = blockIdx.x, tid = threadIdx.x;
    float v[8];
    float m = -3.402823466e38f;
#pragma unroll
    for (int i = 0; i < 8; ++i) {
        v[i] = score[(size_t)b * Tn + tid + i * 256];
        m = fmaxf(m, v[i]);
    }
    red[tid] = m; __syncthreads();
    for (int s = 128; s > 0; s >>= 1) {
        if (tid < s) red[tid] = fmaxf(red[tid], red[tid + s]);
        __syncthreads();
    }
    m = red[0]; __syncthreads();
    float sum = 0.0f;
#pragma unroll
    for (int i = 0; i < 8; ++i) { v[i] = __expf(v[i] - m); sum += v[i]; }
    red[tid] = sum; __syncthreads();
    for (int s = 128; s > 0; s >>= 1) {
        if (tid < s) red[tid] += red[tid + s];
        __syncthreads();
    }
    const float inv = 1.0f / red[0];
#pragma unroll
    for (int i = 0; i < 8; ++i)
        attn[(size_t)b * Tn + tid + i * 256] = v[i] * inv;
}

// ---------------------------------------------------------------------------
// Kernel 5: context[b,f] = sum_t attn[b,t] * features[b,t,f]
// ---------------------------------------------------------------------------
__global__ __launch_bounds__(256)
void context_kernel(const float* __restrict__ features,
                    const float* __restrict__ attn,
                    float* __restrict__ ctx)
{
    __shared__ float w[Tn];
    const int b = blockIdx.x, tid = threadIdx.x;
    const int f = blockIdx.y * 256 + tid;
    for (int i = tid; i < Tn; i += 256) w[i] = attn[(size_t)b * Tn + i];
    __syncthreads();
    const float* fp = features + (size_t)b * Tn * Fn + f;
    float acc = 0.0f;
#pragma unroll 8
    for (int t = 0; t < Tn; ++t)
        acc = fmaf(w[t], fp[(size_t)t * Fn], acc);
    ctx[b * Fn + f] = acc;
}

// ---------------------------------------------------------------------------
extern "C" void kernel_launch(void* const* d_in, const int* in_sizes, int n_in,
                              void* d_out, int out_size, void* d_ws, size_t ws_size,
                              hipStream_t stream)
{
    (void)in_sizes; (void)n_in; (void)out_size; (void)ws_size;
    const float* features = (const float*)d_in[0];   // [B,T,F]
    const float* hidden   = (const float*)d_in[1];   // [B,F]
    const float* W1       = (const float*)d_in[2];   // [F,U]
    const float* b1       = (const float*)d_in[3];   // [U]
    const float* W2       = (const float*)d_in[4];   // [F,U]
    const float* b2       = (const float*)d_in[5];   // [U]
    const float* V        = (const float*)d_in[6];   // [U,1]
    // d_in[7] = bv: cancels inside softmax; not needed.

    float* out  = (float*)d_out;
    float* ctx  = out;                               // [B,F]
    float* attn = out + (size_t)Bn * Fn;             // [B,T,1]

    char* ws = (char*)d_ws;
    _Float16* w1p   = (_Float16*)ws;                               // 512 KB
    float*    projh = (float*)(ws + (size_t)Fn * Un * 2);          // 128 KB
    float*    score = (float*)(ws + (size_t)Fn * Un * 2
                                  + (size_t)Bn * Un * 4);          // 512 KB

    pack_w1_kernel<<<(Fn * Un) / 256, 256, 0, stream>>>(W1, w1p);
    projh_kernel<<<Bn, 256, 0, stream>>>(hidden, W2, b2, projh);
    score_kernel<<<Bn * (Tn / 32), 256, 0, stream>>>(features, w1p, b1, projh, V, score);
    softmax_kernel<<<Bn, 256, 0, stream>>>(score, attn);
    context_kernel<<<dim3(Bn, Fn / 256), 256, 0, stream>>>(features, attn, ctx);
}
